// MultiheadAttention_28681791602715
// MI455X (gfx1250) — matrix operand order repaired and checked by reading, not compiled, not run on hardware
//
#include <hip/hip_runtime.h>
#include <hip/hip_bf16.h>
#include <stdint.h>

// Shapes (fixed by the reference)
#define S_LEN  2048
#define BATCH  4
#define DMODEL 1024
#define NHEAD  16
#define DHEAD  64

typedef __attribute__((ext_vector_type(16))) __bf16 v16bf;
typedef __attribute__((ext_vector_type(8)))  __bf16 v8bf;
typedef __attribute__((ext_vector_type(8)))  float  v8f;
typedef uint32_t u32x4 __attribute__((ext_vector_type(4)));
typedef int32_t  i32x4 __attribute__((ext_vector_type(4)));
typedef int32_t  i32x8 __attribute__((ext_vector_type(8)));

// A-matrix fragment (16x32 bf16): per lane two contiguous 8-elem chunks.
union AFrag {
    v16bf v;
    struct { v8bf lo, hi; } c;
};

__device__ __forceinline__ v8f wmma_bf16(const AFrag& a, const v16bf& b, v8f c) {
    // D = A(16x32 bf16) * B(32x16 bf16) + C(f32)
    return __builtin_amdgcn_wmma_f32_16x16x32_bf16(
        /*neg_a=*/false, a.v, /*neg_b=*/false, b,
        /*c_mod=*/(short)0, c, /*reuse_a=*/false, /*reuse_b=*/false);
}

__device__ __forceinline__ v8f zero8() {
    v8f z = {0.f, 0.f, 0.f, 0.f, 0.f, 0.f, 0.f, 0.f};
    return z;
}

// ---------------------------------------------------------------------------
// Tensor Data Mover: 2D tile Global -> LDS, bf16 (data_size = 2 bytes).
// D# per CDNA5 ISA ch.8: group0 = {count, lds_addr, global_addr[56:0], type=2},
// group1 = {data_size, tensor_dim0/1, tile_dim0/1, tensor_dim0_stride},
// groups 2/3 zero (2D tensor). Tracked with TENSORcnt.
// ---------------------------------------------------------------------------
#if defined(__HIP_DEVICE_COMPILE__) && \
    __has_builtin(__builtin_amdgcn_tensor_load_to_lds) && \
    __has_builtin(__builtin_amdgcn_s_wait_tensorcnt)
#define USE_TDM 1
#else
#define USE_TDM 0
#endif

#if USE_TDM
__device__ __forceinline__ void tdm_load_2d(void* lds_dst, const void* gsrc,
                                            uint32_t tile_w,      // contiguous dim, elems
                                            uint32_t tile_h,      // rows
                                            uint32_t stride_el,   // row stride, elems
                                            uint32_t dim0, uint32_t dim1)
{
    uint64_t ga = (uint64_t)(uintptr_t)gsrc;
    uint32_t la = (uint32_t)(uintptr_t)lds_dst;  // flat LDS addr: low 32 bits = LDS offset
    u32x4 g0;
    g0[0] = 1u;                                   // count=1, user descriptor
    g0[1] = la;                                   // lds_addr (bytes)
    g0[2] = (uint32_t)ga;                         // global_addr[31:0]
    g0[3] = (uint32_t)((ga >> 32) & 0x1FFFFFFu)   // global_addr[56:32]
          | (2u << 30);                           // type=2 ("image")
    i32x8 g1;
    g1[0] = 0x10000;                              // data_size=1 -> 2 bytes
    g1[1] = (int32_t)((dim0 & 0xFFFFu) << 16);                      // tensor_dim0[15:0]
    g1[2] = (int32_t)((dim0 >> 16) | ((dim1 & 0xFFFFu) << 16));     // dim0[31:16]|dim1[15:0]
    g1[3] = (int32_t)((dim1 >> 16) | (tile_w << 16));               // dim1[31:16]|tile_dim0
    g1[4] = (int32_t)tile_h;                      // tile_dim1 (tile_dim2 = 0)
    g1[5] = (int32_t)stride_el;                   // tensor_dim0_stride[31:0]
    g1[6] = 0;                                    // stride0 hi | dim1_stride lo (unused, 2D)
    g1[7] = 0;
    i32x4 z4 = {0, 0, 0, 0};
#if defined(__clang_major__) && (__clang_major__ >= 23)
    i32x8 z8 = {0, 0, 0, 0, 0, 0, 0, 0};
    __builtin_amdgcn_tensor_load_to_lds(g0, g1, z4, z4, z8, 0);
#else
    __builtin_amdgcn_tensor_load_to_lds(g0, g1, z4, z4, 0);
#endif
}
#define TDM_WAIT() __builtin_amdgcn_s_wait_tensorcnt((short)0)
#endif

// ---------------------------------------------------------------------------
// Kernel 0: f32 -> bf16 conversion (x and the four weight matrices)
// ---------------------------------------------------------------------------
__global__ void cvt_f32_bf16(const float* __restrict__ src,
                             __bf16* __restrict__ dst, int n) {
    int i = blockIdx.x * blockDim.x + threadIdx.x;
    if (i < n) dst[i] = (__bf16)src[i];
}

// ---------------------------------------------------------------------------
// Kernel 1: GEMM  Y = X (8192x1024) * W^T (1024x1024 row-major W[d_out][k]) + b
// Block = 4 waves sharing a 32-row A slab (TDM-staged, double-buffered in LDS);
// each wave owns a 64-col stripe -> block tile 32x256. bf16 WMMA, f32 acc.
// mode 0: bf16 scatter to (b,h,s,dh); optional per-head scale from beta (Q)
// mode 1: bf16 scatter to (b,h,dh,s)  (V transposed for the PV WMMA)
// mode 2: f32 row-major (final output projection -> d_out)
// ---------------------------------------------------------------------------
__global__ void __launch_bounds__(128)
gemm_qkvo(const __bf16* __restrict__ X, const __bf16* __restrict__ W,
          const float* __restrict__ bias, const float* __restrict__ beta,
          __bf16* __restrict__ dst_bf, float* __restrict__ dst_f32, int mode)
{
    __shared__ __attribute__((aligned(32))) __bf16 Aslab[2][32 * 32];

    const int tid  = threadIdx.x;
    const int wv   = tid >> 5;
    const int lane = tid & 31;
    const int m16  = lane & 15;
    const int hi   = lane >> 4;

    const int rt = blockIdx.x >> 2;          // 0..255: 32-row tile
    const int cb = blockIdx.x & 3;           // 0..3:   256-col block
    const int row0 = rt * 32;
    const int col0 = cb * 256 + wv * 64;

    v8f acc[2][4];
#pragma unroll
    for (int i = 0; i < 2; i++)
#pragma unroll
        for (int j = 0; j < 4; j++) acc[i][j] = zero8();

#if USE_TDM
    if (wv == 0)   // one DMA per block: 32x32 bf16 slab (2KB), rows of X
        tdm_load_2d(&Aslab[0][0], X + (size_t)row0 * DMODEL,
                    32, 32, DMODEL, DMODEL, 8192);
#endif

    for (int kb = 0; kb < DMODEL; kb += 32) {
        const int cur = (kb >> 5) & 1;
#if USE_TDM
        if (wv == 0) TDM_WAIT();
        __syncthreads();                      // slab[cur] ready; slab[cur^1] free
        if (wv == 0 && kb + 32 < DMODEL)      // prefetch next slab during compute
            tdm_load_2d(&Aslab[cur ^ 1][0],
                        X + (size_t)row0 * DMODEL + kb + 32,
                        32, 32, DMODEL, DMODEL, 8192);
#else
        __syncthreads();                      // all reads of slab[cur] done
        {   // cooperative copy: 128 threads x 16B = 2KB
            const int r = tid >> 2, c = (tid & 3) * 8;
            *(v8bf*)&Aslab[cur][r * 32 + c] =
                *(const v8bf*)(X + (size_t)(row0 + r) * DMODEL + kb + c);
        }
        __syncthreads();
#endif
        AFrag a[2];
#pragma unroll
        for (int i = 0; i < 2; i++) {
            a[i].c.lo = *(const v8bf*)&Aslab[cur][(i*16 + m16) * 32 + hi*16];
            a[i].c.hi = *(const v8bf*)&Aslab[cur][(i*16 + m16) * 32 + hi*16 + 8];
        }
#pragma unroll
        for (int j = 0; j < 4; j++) {
            const __bf16* pw = W + (size_t)(col0 + j*16 + m16) * DMODEL + kb + hi*16;
            v16bf bw = *(const v16bf*)pw;
            acc[0][j] = wmma_bf16(a[0], bw, acc[0][j]);
            acc[1][j] = wmma_bf16(a[1], bw, acc[1][j]);
        }
    }

    // Epilogue. C layout: lane holds col n = m16, rows r + 8*hi.
#pragma unroll
    for (int j = 0; j < 4; j++) {
        const int col = col0 + j*16 + m16;
        const float bc = bias[col];
        float scale = 1.0f;
        const int h  = col >> 6;
        const int dh = col & 63;
        if (mode == 0 && beta) scale = __expf(-beta[h]) * 0.125f;  // 1/(8*exp(beta))
#pragma unroll
        for (int i = 0; i < 2; i++) {
#pragma unroll
            for (int r = 0; r < 8; r++) {
                const int row = row0 + i*16 + r + 8*hi;   // n = s*BATCH + b
                float val = acc[i][j][r] + bc;
                if (mode == 2) {
                    dst_f32[(size_t)row * DMODEL + col] = val;
                } else {
                    const int s  = row >> 2;      // BATCH == 4
                    const int bb = row & 3;
                    if (mode == 0) {
                        dst_bf[(((size_t)(bb*NHEAD + h)) * S_LEN + s) * DHEAD + dh] =
                            (__bf16)(val * scale);
                    } else {
                        dst_bf[(((size_t)(bb*NHEAD + h)) * DHEAD + dh) * S_LEN + s] =
                            (__bf16)val;
                    }
                }
            }
        }
    }
}

// ---------------------------------------------------------------------------
// Kernel 2: flash attention, one wave per (b, h, 16-row q tile).
// Q: (b,h,s,dh) bf16 (pre-scaled), K: (b,h,s,dh) bf16, Vt: (b,h,dh,s) bf16.
// K/V tiles are TDM-DMA'd into double-buffered LDS: the DMA for iteration n+1
// overlaps the 8 WMMAs + softmax of iteration n (TENSORcnt pipelining).
// Output written bf16 to (s*B+b, d) for the final projection GEMM.
// ---------------------------------------------------------------------------
__global__ void __launch_bounds__(32)
attn_flash(const __bf16* __restrict__ Q, const __bf16* __restrict__ K,
           const __bf16* __restrict__ Vt, __bf16* __restrict__ dstA)
{
#if USE_TDM
    __shared__ __attribute__((aligned(32))) __bf16 Kslab[2][32 * 64]; // 32 keys x 64 dh
    __shared__ __attribute__((aligned(32))) __bf16 Vslab[2][64 * 32]; // 64 dh x 32 s
#endif
    __shared__ __attribute__((aligned(32))) __bf16 Pbuf[16 * 32];

    const int lane = threadIdx.x;
    const int m16  = lane & 15;
    const int hi   = lane >> 4;
    const int qt = blockIdx.x;      // 0..127
    const int h  = blockIdx.y;      // 0..15
    const int bb = blockIdx.z;      // 0..3

    const size_t bh = (size_t)bb * NHEAD + h;
    const __bf16* Qp = Q  + (bh * S_LEN + (size_t)qt * 16) * DHEAD;
    const __bf16* Kp = K  + bh * S_LEN * DHEAD;
    const __bf16* Vp = Vt + bh * DHEAD * (size_t)S_LEN;

    // Q fragments (16x64 = two 16x32 A-frags), held for the whole k-loop.
    AFrag aQ[2];
    {
        const __bf16* p = Qp + (size_t)m16 * DHEAD;
        aQ[0].c.lo = *(const v8bf*)(p + hi*16);
        aQ[0].c.hi = *(const v8bf*)(p + hi*16 + 8);
        aQ[1].c.lo = *(const v8bf*)(p + 32 + hi*16);
        aQ[1].c.hi = *(const v8bf*)(p + 32 + hi*16 + 8);
    }

    v8f O[4];
#pragma unroll
    for (int cg = 0; cg < 4; cg++) O[cg] = zero8();
    float mrun[8], lrun[8];
#pragma unroll
    for (int r = 0; r < 8; r++) { mrun[r] = -3.0e38f; lrun[r] = 0.f; }

#if USE_TDM
    // Prologue DMA: first K/V tiles into buffer 0.
    tdm_load_2d(&Kslab[0][0], Kp, DHEAD, 32, DHEAD, DHEAD, S_LEN);
    tdm_load_2d(&Vslab[0][0], Vp, 32, DHEAD, S_LEN, S_LEN, DHEAD);
#endif

    for (int k0 = 0; k0 < S_LEN; k0 += 32) {
#if USE_TDM
        const int cur = (k0 >> 5) & 1;
        TDM_WAIT();                            // current K/V tiles landed in LDS
        if (k0 + 32 < S_LEN) {                 // async prefetch of next tiles
            tdm_load_2d(&Kslab[cur ^ 1][0], Kp + (size_t)(k0 + 32) * DHEAD,
                        DHEAD, 32, DHEAD, DHEAD, S_LEN);
            tdm_load_2d(&Vslab[cur ^ 1][0], Vp + (k0 + 32),
                        32, DHEAD, S_LEN, S_LEN, DHEAD);
        }
        const __bf16* kr0 = &Kslab[cur][(size_t)m16 * DHEAD + hi*16];
        const __bf16* kr1 = &Kslab[cur][(size_t)(16 + m16) * DHEAD + hi*16];
#else
        const __bf16* kr0 = Kp + (size_t)(k0 + m16) * DHEAD + hi*16;
        const __bf16* kr1 = Kp + (size_t)(k0 + 16 + m16) * DHEAD + hi*16;
#endif
        v16bf bk00 = *(const v16bf*)(kr0);
        v16bf bk01 = *(const v16bf*)(kr0 + 32);
        v16bf bk10 = *(const v16bf*)(kr1);
        v16bf bk11 = *(const v16bf*)(kr1 + 32);

        v8f s0 = wmma_bf16(aQ[1], bk01, wmma_bf16(aQ[0], bk00, zero8()));
        v8f s1 = wmma_bf16(aQ[1], bk11, wmma_bf16(aQ[0], bk10, zero8()));

        // Online softmax. Row r+8*hi lives in one 16-lane half -> width-16 shuffles.
        float p0[8], p1[8], alpha[8];
#pragma unroll
        for (int r = 0; r < 8; r++) {
            float tm = fmaxf(s0[r], s1[r]);
            tm = fmaxf(tm, __shfl_xor(tm, 1, 16));
            tm = fmaxf(tm, __shfl_xor(tm, 2, 16));
            tm = fmaxf(tm, __shfl_xor(tm, 4, 16));
            tm = fmaxf(tm, __shfl_xor(tm, 8, 16));
            const float nm = fmaxf(mrun[r], tm);
            alpha[r] = __expf(mrun[r] - nm);
            mrun[r]  = nm;
            p0[r] = __expf(s0[r] - nm);
            p1[r] = __expf(s1[r] - nm);
            float rs = p0[r] + p1[r];
            rs += __shfl_xor(rs, 1, 16);
            rs += __shfl_xor(rs, 2, 16);
            rs += __shfl_xor(rs, 4, 16);
            rs += __shfl_xor(rs, 8, 16);
            lrun[r] = lrun[r] * alpha[r] + rs;
        }
#pragma unroll
        for (int cg = 0; cg < 4; cg++)
#pragma unroll
            for (int r = 0; r < 8; r++) O[cg][r] *= alpha[r];

        // C-layout -> A-layout transpose of P through LDS (16x32 bf16).
#pragma unroll
        for (int r = 0; r < 8; r++) {
            Pbuf[(r + 8*hi) * 32 + m16]      = (__bf16)p0[r];
            Pbuf[(r + 8*hi) * 32 + 16 + m16] = (__bf16)p1[r];
        }
        __syncthreads();
        AFrag aP;
        aP.c.lo = *(const v8bf*)(&Pbuf[m16 * 32 + hi*16]);
        aP.c.hi = *(const v8bf*)(&Pbuf[m16 * 32 + hi*16 + 8]);
        __syncthreads();

        // O(16x64) += P(16x32) x V(32x64)
#pragma unroll
        for (int cg = 0; cg < 4; cg++) {
#if USE_TDM
            v16bf bv = *(const v16bf*)&Vslab[cur][(cg*16 + m16) * 32 + hi*16];
#else
            v16bf bv = *(const v16bf*)(Vp + (size_t)(cg*16 + m16) * S_LEN + k0 + hi*16);
#endif
            O[cg] = wmma_bf16(aP, bv, O[cg]);
        }
    }

    // Normalize and scatter to (s*B + b, d) bf16 for the output projection.
#pragma unroll
    for (int cg = 0; cg < 4; cg++) {
#pragma unroll
        for (int r = 0; r < 8; r++) {
            const int srow = qt * 16 + r + 8*hi;
            const int d    = h * DHEAD + cg * 16 + m16;
            const float inv = 1.0f / lrun[r];
            dstA[((size_t)srow * BATCH + bb) * DMODEL + d] = (__bf16)(O[cg][r] * inv);
        }
    }
}

// ---------------------------------------------------------------------------
// Host launcher
// ---------------------------------------------------------------------------
extern "C" void kernel_launch(void* const* d_in, const int* in_sizes, int n_in,
                              void* d_out, int out_size, void* d_ws, size_t ws_size,
                              hipStream_t stream) {
    const float* x    = (const float*)d_in[0];
    const float* Wq   = (const float*)d_in[1];
    const float* bq   = (const float*)d_in[2];
    const float* Wk   = (const float*)d_in[3];
    const float* bk   = (const float*)d_in[4];
    const float* Wv   = (const float*)d_in[5];
    const float* bv   = (const float*)d_in[6];
    const float* Wo   = (const float*)d_in[7];
    const float* bo   = (const float*)d_in[8];
    const float* beta = (const float*)d_in[9];
    float* out = (float*)d_out;

    char* ws = (char*)d_ws;
    const size_t MB = 1024ull * 1024ull;
    __bf16* xb  = (__bf16*)(ws + 0);          // 16 MiB: x as bf16, (s*B+b, d)
    __bf16* wqb = (__bf16*)(ws + 16 * MB);    //  2 MiB
    __bf16* wkb = (__bf16*)(ws + 18 * MB);
    __bf16* wvb = (__bf16*)(ws + 20 * MB);
    __bf16* wob = (__bf16*)(ws + 22 * MB);
    __bf16* Qb  = (__bf16*)(ws + 24 * MB);    // 16 MiB: (b,h,s,dh), pre-scaled
    __bf16* Kb  = (__bf16*)(ws + 40 * MB);    // 16 MiB: (b,h,s,dh)
    __bf16* Vtb = (__bf16*)(ws + 56 * MB);    // 16 MiB: (b,h,dh,s)
    __bf16* Ab  = (__bf16*)(ws + 72 * MB);    // 16 MiB: attn out, (s*B+b, d)

    const int nx = S_LEN * BATCH * DMODEL;    // 8,388,608
    const int nw = DMODEL * DMODEL;           // 1,048,576
    cvt_f32_bf16<<<(nx + 255) / 256, 256, 0, stream>>>(x,  xb,  nx);
    cvt_f32_bf16<<<(nw + 255) / 256, 256, 0, stream>>>(Wq, wqb, nw);
    cvt_f32_bf16<<<(nw + 255) / 256, 256, 0, stream>>>(Wk, wkb, nw);
    cvt_f32_bf16<<<(nw + 255) / 256, 256, 0, stream>>>(Wv, wvb, nw);
    cvt_f32_bf16<<<(nw + 255) / 256, 256, 0, stream>>>(Wo, wob, nw);

    // 1024 blocks x 128 threads: 256 row-tiles x 4 col-blocks (4 waves x 64 cols).
    gemm_qkvo<<<1024, 128, 0, stream>>>(xb, wqb, bq, beta,    Qb,  nullptr, 0);
    gemm_qkvo<<<1024, 128, 0, stream>>>(xb, wkb, bk, nullptr, Kb,  nullptr, 0);
    gemm_qkvo<<<1024, 128, 0, stream>>>(xb, wvb, bv, nullptr, Vtb, nullptr, 1);

    attn_flash<<<dim3(S_LEN / 16, NHEAD, BATCH), 32, 0, stream>>>(Qb, Kb, Vtb, Ab);

    gemm_qkvo<<<1024, 128, 0, stream>>>(Ab, wob, bo, nullptr, nullptr, out, 2);
}
